// MinkResNet_41798621725225
// MI455X (gfx1250) — matrix operand order repaired and checked by reading, not compiled, not run on hardware
//
#include <hip/hip_runtime.h>

// ---------------------------------------------------------------------------
// CDNA5 (gfx1250) sparse-ResNet forward:
//   gather -> per-offset GEMM (v_wmma_f32_16x16x32_bf16) -> scatter-add.
//   Weight panels staged to LDS by the Tensor Data Mover (TDM), double
//   buffered, synchronized with s_wait_tensorcnt. BN stats/apply epilogues.
// Block = 8 waves: (offset k, 128-row M panel, 64-col N panel).
// Wave   = 16x64 output strip: 4 x v_wmma_f32_16x16x32_bf16 per K-chunk.
// ---------------------------------------------------------------------------

typedef __attribute__((ext_vector_type(16))) __bf16   v16bf;
typedef __attribute__((ext_vector_type(8)))  float    v8f;
typedef __attribute__((ext_vector_type(8)))  unsigned v8u;
typedef __attribute__((ext_vector_type(4)))  unsigned tdm_u4;
typedef __attribute__((ext_vector_type(8)))  int      tdm_i8;
typedef __attribute__((ext_vector_type(4)))  int      tdm_i4;

// Issue a TDM 2D tile load: W[k][kc:kc+32][n0:n0+64] (f32) -> LDS buffer.
// D# per ISA ch.8: group0 = {count/flags, lds_addr, global_addr, type},
// group1 = {data_size, tensor_dim0/1, tile_dim0/1/2, tensor_dim0_stride}.
// OOB reads return zero => free padding for Cout<64 / Cin<32 edges.
__device__ __forceinline__ void tdm_stage_w(const float* gsrc, void* ldsdst,
                                            int rem_cols, int rem_rows, int ld)
{
    unsigned long long ga = (unsigned long long)(size_t)gsrc;
    unsigned ldsoff = (unsigned)(unsigned long long)(size_t)ldsdst;

    tdm_u4 g0;
    g0[0] = 1u;                                   // count=1, user mode, no gather
    g0[1] = ldsoff;                               // lds_addr (bytes)
    g0[2] = (unsigned)(ga & 0xFFFFFFFFull);       // global_addr[31:0]
    g0[3] = (unsigned)((ga >> 32) & 0x01FFFFFFull) | 0x80000000u; // addr[56:32] | type=2

    unsigned td0 = (unsigned)rem_cols;            // tensor_dim0 (OOB pad -> 0)
    unsigned td1 = (unsigned)rem_rows;            // tensor_dim1
    tdm_i8 g1;
    g1[0] = 0x20000;                                               // data_size=4B
    g1[1] = (int)((td0 & 0xFFFFu) << 16);                          // tensor_dim0 lo
    g1[2] = (int)(((td0 >> 16) & 0xFFFFu) | ((td1 & 0xFFFFu) << 16));
    g1[3] = (int)(((td1 >> 16) & 0xFFFFu) | (64u << 16));          // tile_dim0=64
    g1[4] = 32;                                                    // tile_dim1=32, tile_dim2=0
    g1[5] = ld;                                                    // tensor_dim0_stride[31:0]
    g1[6] = 0;
    g1[7] = 0;
    tdm_i4 gz = {0, 0, 0, 0};
#if __clang_major__ >= 23
    tdm_i8 gz8 = {0, 0, 0, 0, 0, 0, 0, 0};
    __builtin_amdgcn_tensor_load_to_lds(g0, g1, gz, gz, gz8, 0);
#else
    __builtin_amdgcn_tensor_load_to_lds(g0, g1, gz, gz, 0);
#endif
}

// ISA 16-bit A 16x32 layout: lane l holds row m=l&15; VGPR v<4 -> K=kb+2v,
// kb=(l<16)?0:8 ; v>=4 -> K=16+kb+2(v-4). (consecutive K pairs -> u32 loads)
// B 32x16: lane l holds col n=l&15, kb=(l<16)?0:8; e<8 -> K=kb+e, e>=8 -> K=16+kb+(e-8).
// D f32 16x16: lane l col n=l&15, VGPR r -> row m=r+((l>=16)?8:0).
__global__ void __launch_bounds__(256)
sconv_wmma(const __bf16* __restrict__ X, const float* __restrict__ W,
           const int* __restrict__ in_idx, const int* __restrict__ out_idx,
           float* __restrict__ Y, int K, int M, int Cin, int Cout, int atomic)
{
    __shared__ float lds_w[2][32 * 64];           // double-buffered weight panel

    const int ntiles = (Cout + 63) >> 6;          // 64-col N panels
    const int mtiles = (M + 127) >> 7;            // 128-row M panels
    int blk = blockIdx.x;
    int k   = blk / (mtiles * ntiles);
    int rem = blk - k * mtiles * ntiles;
    int mp  = rem / ntiles;
    int np  = rem - mp * ntiles;
    const int n0 = np << 6;

    const int waveid = threadIdx.x >> 5;
    const int lane   = threadIdx.x & 31;
    const int m0     = (mp * 8 + waveid) << 4;

    // Gathered A row for this lane (clamped; invalid rows never stored).
    int ma  = m0 + (lane & 15);
    int mac = (ma < M) ? ma : (M - 1);
    int ra  = in_idx ? in_idx[(size_t)k * M + mac] : mac;
    const __bf16* xrow = X + (size_t)ra * Cin;

    const int kbA = (lane < 16) ? 0 : 8;
    const int kbB = (lane < 16) ? 0 : 8;
    const float* wk = W + (size_t)k * Cin * Cout;

    v8f acc[4] = {};
    int cur = 0;
    if (threadIdx.x == 0)
        tdm_stage_w(wk + (size_t)0 * Cout + n0, &lds_w[0][0], Cout - n0, Cin, Cout);

    for (int kc = 0; kc < Cin; kc += 32) {
        __builtin_amdgcn_s_wait_tensorcnt((short)0);   // panel [cur] landed
        __syncthreads();
        if (kc + 32 < Cin && threadIdx.x == 0)         // overlap next DMA
            tdm_stage_w(wk + (size_t)(kc + 32) * Cout + n0, &lds_w[cur ^ 1][0],
                        Cout - n0, Cin - kc - 32, Cout);

        // ---- A tile (bf16 pairs; fast aligned u32 loads for full chunks) ----
        v8u au;
        if (Cin - kc >= 32) {
            if (kc + 32 < Cin) __builtin_prefetch(xrow + kc + 32, 0, 3);
#pragma unroll
            for (int v = 0; v < 8; ++v) {
                int kk = kc + ((v < 4) ? (kbA + 2 * v) : (16 + kbA + 2 * (v - 4)));
                au[v] = *reinterpret_cast<const unsigned*>(xrow + kk);
            }
        } else {
#pragma unroll
            for (int v = 0; v < 8; ++v) {
                int kk = kc + ((v < 4) ? (kbA + 2 * v) : (16 + kbA + 2 * (v - 4)));
                unsigned lo = (kk     < Cin) ? (unsigned)__builtin_bit_cast(unsigned short, xrow[kk])     : 0u;
                unsigned hi = (kk + 1 < Cin) ? (unsigned)__builtin_bit_cast(unsigned short, xrow[kk + 1]) : 0u;
                au[v] = lo | (hi << 16);
            }
        }
        v16bf a = __builtin_bit_cast(v16bf, au);

        // ---- 4 N-subtiles: B from LDS (f32 -> bf16), reuse A ----
        const float* lw = &lds_w[cur][0];
#pragma unroll
        for (int nb = 0; nb < 4; ++nb) {
            v16bf b;
#pragma unroll
            for (int e = 0; e < 16; ++e)
                b[e] = (__bf16)lw[((e < 8) ? (kbB + e) : (16 + kbB + (e - 8))) * 64 + nb * 16 + (lane & 15)];
            acc[nb] = __builtin_amdgcn_wmma_f32_16x16x32_bf16(
                false, a, false, b, (short)0, acc[nb], false, false);
        }
        __syncthreads();                               // reads done before refill
        cur ^= 1;
    }

    // ---- scatter 16x64 strip ----
    const int mofs = (lane >= 16) ? 8 : 0;
#pragma unroll
    for (int nb = 0; nb < 4; ++nb) {
        int col = n0 + nb * 16 + (lane & 15);
        if (col < Cout) {
#pragma unroll
            for (int r = 0; r < 8; ++r) {
                int mm = m0 + r + mofs;
                if (mm < M) {
                    int row = out_idx ? out_idx[(size_t)k * M + mm] : mm;
                    float* p = Y + (size_t)row * Cout + col;
                    if (atomic) atomicAdd(p, acc[nb][r]);
                    else        *p = acc[nb][r];
                }
            }
        }
    }
}

// Per-channel mean / rsqrt(var+eps) over the point dimension (one block/channel).
__global__ void __launch_bounds__(256)
bn_stats(const float* __restrict__ Y, int n, int C, float* __restrict__ st)
{
    __shared__ float ss[256], ss2[256];
    const int c = blockIdx.x;
    float s = 0.f, s2 = 0.f;
    for (int i = threadIdx.x; i < n; i += blockDim.x) {
        float v = Y[(size_t)i * C + c];
        s += v; s2 += v * v;
    }
    ss[threadIdx.x] = s; ss2[threadIdx.x] = s2;
    __syncthreads();
    for (int w = 128; w > 0; w >>= 1) {
        if ((int)threadIdx.x < w) {
            ss[threadIdx.x]  += ss[threadIdx.x + w];
            ss2[threadIdx.x] += ss2[threadIdx.x + w];
        }
        __syncthreads();
    }
    if (threadIdx.x == 0) {
        float m = ss[0] / (float)n;
        float v = ss2[0] / (float)n - m * m;
        st[c]     = m;
        st[C + c] = rsqrtf(v + 1e-5f);
    }
}

// Fused epilogue: bn(Y) [+ bn(Y2)] [+ residual(bf16)] [relu] -> bf16 out.
__global__ void __launch_bounds__(256)
bn_apply(const float* __restrict__ Y, const float* __restrict__ st,
         const float* __restrict__ g, const float* __restrict__ b,
         const float* __restrict__ Y2, const float* __restrict__ st2,
         const float* __restrict__ g2, const float* __restrict__ b2,
         const __bf16* __restrict__ res, int relu, int n, int C,
         __bf16* __restrict__ out)
{
    long idx = (long)blockIdx.x * blockDim.x + threadIdx.x;
    long tot = (long)n * C;
    if (idx >= tot) return;
    int c = (int)(idx % C);
    float v = (Y[idx] - st[c]) * st[C + c] * g[c] + b[c];
    if (Y2)  v += (Y2[idx] - st2[c]) * st2[C + c] * g2[c] + b2[c];
    if (res) v += (float)res[idx];
    if (relu) v = fmaxf(v, 0.0f);
    out[idx] = (__bf16)v;
}

__global__ void __launch_bounds__(256)
f32_to_bf16(const float* __restrict__ in, __bf16* __restrict__ out, long n)
{
    long i = (long)blockIdx.x * blockDim.x + threadIdx.x;
    if (i < n) out[i] = (__bf16)in[i];
}

__global__ void __launch_bounds__(256)
bias_out(const float* __restrict__ acc, const float* __restrict__ bias,
         float* __restrict__ out, int n, int C)
{
    long i = (long)blockIdx.x * blockDim.x + threadIdx.x;
    if (i < (long)n * C) out[i] = acc[i] + bias[i % C];
}

__global__ void __launch_bounds__(256)
voxel_center_k(const int* __restrict__ coords, float* __restrict__ out, int n)
{
    int i = blockIdx.x * blockDim.x + threadIdx.x;
    if (i < n * 3) {
        int r = i / 3, c = i - r * 3;
        out[i] = (float)coords[r * 4 + c] + 8.0f;
    }
}

// ---------------------------------------------------------------------------
extern "C" void kernel_launch(void* const* d_in, const int* in_sizes, int n_in,
                              void* d_out, int out_size, void* d_ws, size_t ws_size,
                              hipStream_t stream)
{
    (void)in_sizes; (void)n_in; (void)out_size; (void)ws_size;

    const int CS[5]  = {64, 128, 256, 512, 512};
    const int BR[4]  = {3, 4, 6, 3};
    const int NV[5]  = {30000, 15000, 8000, 4000, 2000};
    const int M27[5] = {18000, 9000, 4800, 2400, 1200};
    const int MDS[4] = {3750, 1875, 1000, 500};

    const float* feats   = (const float*)d_in[0];
    const int*   coords4 = (const int*)d_in[1];
    const int *km_in[5], *km_out[5], *dsi_in[4], *dsi_out[4];
    for (int l = 0; l < 5; ++l) { km_in[l] = (const int*)d_in[2 + 2 * l]; km_out[l] = (const int*)d_in[3 + 2 * l]; }
    for (int i = 0; i < 4; ++i) { dsi_in[i] = (const int*)d_in[12 + 2 * i]; dsi_out[i] = (const int*)d_in[13 + 2 * i]; }
    const float* P = (const float*)d_in[20];

    // ---- param offsets: jax pytree flatten order (dict keys alphabetical) ----
    // top: cls_b, cls_w, ds, stages, stem ; conv_p: b, g, w ; block: c1, c2, dsp
    size_t off = 0;
    auto take = [&](size_t n) { size_t o = off; off += n; return o; };
    struct CP { size_t b, g, w; };
    size_t cls_b_o = take(20);
    size_t cls_w_o = take((size_t)512 * 20);
    size_t ds_b[4], ds_g[4], ds_w[4];
    for (int i = 0; i < 4; ++i) {
        ds_b[i] = take(CS[i]); ds_g[i] = take(CS[i]); ds_w[i] = take((size_t)8 * CS[i] * CS[i]);
    }
    CP c1[4][6], c2[4][6], dsp[4];
    for (int i = 0; i < 4; ++i) {
        for (int b = 0; b < BR[i]; ++b) {
            int inc = CS[i], outc = (b < BR[i] - 1) ? CS[i] : CS[i + 1];
            c1[i][b].b = take(outc); c1[i][b].g = take(outc); c1[i][b].w = take((size_t)27 * inc * outc);
            c2[i][b].b = take(outc); c2[i][b].g = take(outc); c2[i][b].w = take((size_t)27 * outc * outc);
            if (inc != outc) { dsp[i].b = take(outc); dsp[i].g = take(outc); dsp[i].w = take((size_t)inc * outc); }
        }
    }
    CP stemp[2];
    stemp[0].b = take(64); stemp[0].g = take(64); stemp[0].w = take((size_t)27 * 3 * 64);
    stemp[1].b = take(64); stemp[1].g = take(64); stemp[1].w = take((size_t)27 * 64 * 64);

    // ---- workspace (bump allocator) ----
    const size_t MAXE = 2100000;   // max n*C over all levels (8000*256 = 2.048M)
    char* wsp = (char*)d_ws;
    auto walloc = [&](size_t bytes) -> void* {
        void* p = (void*)wsp; wsp += (bytes + 255) & ~(size_t)255; return p;
    };
    float*  accA = (float*)walloc(MAXE * 4);
    float*  accB = (float*)walloc(MAXE * 4);
    __bf16* xb[3];
    for (int i = 0; i < 3; ++i) xb[i] = (__bf16*)walloc(MAXE * 2);
    float* stA = (float*)walloc((size_t)2 * 512 * 4);
    float* stB = (float*)walloc((size_t)2 * 512 * 4);

    auto zero = [&](float* p, size_t n) { hipMemsetAsync(p, 0, n * sizeof(float), stream); };
    auto sconv = [&](const __bf16* X, size_t w_o, const int* ii, const int* io,
                     float* Y, int K, int M, int Cin, int Cout, int atomic) {
        int blocks = K * ((M + 127) / 128) * ((Cout + 63) / 64);
        sconv_wmma<<<blocks, 256, 0, stream>>>(X, P + w_o, ii, io, Y, K, M, Cin, Cout, atomic);
    };
    auto stats = [&](const float* Y, int n, int C, float* st) {
        bn_stats<<<C, 256, 0, stream>>>(Y, n, C, st);
    };
    auto apply = [&](const float* Y, const float* st, size_t g_o, size_t b_o,
                     const float* Y2, const float* st2, const float* g2, const float* b2,
                     const __bf16* res, int relu, int n, int C, __bf16* out) {
        long tot = (long)n * C;
        bn_apply<<<(int)((tot + 255) / 256), 256, 0, stream>>>(
            Y, st, P + g_o, P + b_o, Y2, st2, g2, b2, res, relu, n, C, out);
    };

    // ---- feats -> bf16 ----
    { long tot = (long)NV[0] * 3;
      f32_to_bf16<<<(int)((tot + 255) / 256), 256, 0, stream>>>(feats, xb[0], tot); }

    // ---- stem (two 27-offset convs at level 0) ----
    zero(accA, (size_t)NV[0] * 64);
    sconv(xb[0], stemp[0].w, km_in[0], km_out[0], accA, 27, M27[0], 3, 64, 1);
    stats(accA, NV[0], 64, stA);
    apply(accA, stA, stemp[0].g, stemp[0].b, nullptr, nullptr, nullptr, nullptr, nullptr, 1, NV[0], 64, xb[1]);
    zero(accA, (size_t)NV[0] * 64);
    sconv(xb[1], stemp[1].w, km_in[0], km_out[0], accA, 27, M27[0], 64, 64, 1);
    stats(accA, NV[0], 64, stA);
    apply(accA, stA, stemp[1].g, stemp[1].b, nullptr, nullptr, nullptr, nullptr, nullptr, 1, NV[0], 64, xb[0]);
    int ix = 0;

    // ---- 4 stages ----
    for (int i = 0; i < 4; ++i) {
        const int n = NV[i + 1];
        // strided downsample conv (8 offsets)
        zero(accA, (size_t)n * CS[i]);
        sconv(xb[ix], ds_w[i], dsi_in[i], dsi_out[i], accA, 8, MDS[i], CS[i], CS[i], 1);
        stats(accA, n, CS[i], stA);
        int nx = (ix + 1) % 3;
        apply(accA, stA, ds_g[i], ds_b[i], nullptr, nullptr, nullptr, nullptr, nullptr, 1, n, CS[i], xb[nx]);
        ix = nx;
        int curC = CS[i];

        for (int b = 0; b < BR[i]; ++b) {
            const int outc = (b < BR[i] - 1) ? CS[i] : CS[i + 1];
            const int hx = (ix + 1) % 3, ox = (ix + 2) % 3;
            // c1: conv + BN + ReLU
            zero(accA, (size_t)n * outc);
            sconv(xb[ix], c1[i][b].w, km_in[i + 1], km_out[i + 1], accA, 27, M27[i + 1], curC, outc, 1);
            stats(accA, n, outc, stA);
            apply(accA, stA, c1[i][b].g, c1[i][b].b, nullptr, nullptr, nullptr, nullptr, nullptr, 1, n, outc, xb[hx]);
            // c2: conv + BN (no relu yet)
            zero(accA, (size_t)n * outc);
            sconv(xb[hx], c2[i][b].w, km_in[i + 1], km_out[i + 1], accA, 27, M27[i + 1], outc, outc, 1);
            stats(accA, n, outc, stA);
            if (outc != curC) {
                // 1x1 shortcut: dense GEMM + BN, fused into final add+relu
                sconv(xb[ix], dsp[i].w, nullptr, nullptr, accB, 1, n, curC, outc, 0);
                stats(accB, n, outc, stB);
                apply(accA, stA, c2[i][b].g, c2[i][b].b, accB, stB, P + dsp[i].g, P + dsp[i].b,
                      nullptr, 1, n, outc, xb[ox]);
            } else {
                apply(accA, stA, c2[i][b].g, c2[i][b].b, nullptr, nullptr, nullptr, nullptr,
                      xb[ix], 1, n, outc, xb[ox]);
            }
            ix = ox; curC = outc;
        }
    }

    // ---- classifier: [2000x512] @ [512x20] + bias ----
    sconv(xb[ix], cls_w_o, nullptr, nullptr, accA, 1, NV[4], 512, 20, 0);
    { long tot = (long)NV[4] * 20;
      bias_out<<<(int)((tot + 255) / 256), 256, 0, stream>>>(accA, P + cls_b_o, (float*)d_out, NV[4], 20); }

    // ---- voxel centers: coords[:, :3] + 8.0 ----
    { int tot = NV[4] * 3;
      voxel_center_k<<<(tot + 255) / 256, 256, 0, stream>>>(coords4, (float*)d_out + (size_t)NV[4] * 20, NV[4]); }
}
